// Memory_90031104459200
// MI455X (gfx1250) — matrix operand order repaired and checked by reading, not compiled, not run on hardware
//
#include <hip/hip_runtime.h>
#include <hip/hip_fp16.h>

typedef __attribute__((ext_vector_type(16))) _Float16 v16h;
typedef __attribute__((ext_vector_type(8)))  _Float16 v8h;
typedef __attribute__((ext_vector_type(8)))  float    v8f;

#define DIM 256
#define L2EPS 1e-12f

union AF { v16h h; uint4 u[2]; };

__device__ __forceinline__ float wave_reduce_add(float x) {
  #pragma unroll
  for (int off = 16; off > 0; off >>= 1) x += __shfl_xor(x, off, 32);
  return x;
}

// ---------------------------------------------------------------------------
// Pass A: per-row L2 normalize of feat -> f16 buffer, plus f32 scatter-add
// of normalized rows into per-class sums (segment_sum) and counts.
// One wave (32 lanes) per row; each lane owns 8 of the 256 elements.
// ---------------------------------------------------------------------------
__global__ __launch_bounds__(256)
void k_normalize_scatter(const float* __restrict__ feat,
                         const int* __restrict__ label,
                         _Float16* __restrict__ featn_h,
                         float* __restrict__ sums,
                         float* __restrict__ counts,
                         int N) {
  int wave = (int)((blockIdx.x * blockDim.x + threadIdx.x) >> 5);
  int lane = threadIdx.x & 31;
  if (wave >= N) return;

  const float4* src = (const float4*)(feat + (size_t)wave * DIM) + lane * 2;
  float4 a0 = src[0];
  float4 a1 = src[1];
  float x[8] = {a0.x, a0.y, a0.z, a0.w, a1.x, a1.y, a1.z, a1.w};

  float ss = 0.f;
  #pragma unroll
  for (int j = 0; j < 8; ++j) ss = fmaf(x[j], x[j], ss);
  ss = wave_reduce_add(ss);
  float inv = 1.0f / fmaxf(sqrtf(ss), L2EPS);

  v8h h;
  #pragma unroll
  for (int j = 0; j < 8; ++j) {
    x[j] *= inv;
    h[j] = (_Float16)x[j];
  }
  *(v8h*)(featn_h + (size_t)wave * DIM + lane * 8) = h;

  int lbl = label[wave];
  float* dst = sums + (size_t)lbl * DIM + lane * 8;
  #pragma unroll
  for (int j = 0; j < 8; ++j) atomicAdd(dst + j, x[j]);
  if (lane == 0) atomicAdd(counts + lbl, 1.0f);
}

// ---------------------------------------------------------------------------
// Pass B: per-class center update. One wave per class row (padded to Cpad).
//   bc   = l2norm(sums) * (count>0)
//   w    = 1 - (1 - dot(memory, bc)) * (count>0)
//   newm = l2norm(w*memory + (1-w)*bc)   -> f16 (pad rows -> 0)
// ---------------------------------------------------------------------------
__global__ __launch_bounds__(256)
void k_center_update(const float* __restrict__ memory,
                     const float* __restrict__ sums,
                     const float* __restrict__ counts,
                     _Float16* __restrict__ newmem_h,
                     int C, int Cpad) {
  int wave = (int)((blockIdx.x * blockDim.x + threadIdx.x) >> 5);
  int lane = threadIdx.x & 31;
  if (wave >= Cpad) return;

  _Float16* out = newmem_h + (size_t)wave * DIM + lane * 8;
  if (wave >= C) {                 // zero padding rows (wave-uniform branch)
    v8h z = {};
    *(v8h*)out = z;
    return;
  }

  const float4* sp = (const float4*)(sums + (size_t)wave * DIM) + lane * 2;
  const float4* mp = (const float4*)(memory + (size_t)wave * DIM) + lane * 2;
  float4 s0 = sp[0], s1 = sp[1], m0 = mp[0], m1 = mp[1];
  float s[8] = {s0.x, s0.y, s0.z, s0.w, s1.x, s1.y, s1.z, s1.w};
  float m[8] = {m0.x, m0.y, m0.z, m0.w, m1.x, m1.y, m1.z, m1.w};

  float ss = 0.f;
  #pragma unroll
  for (int j = 0; j < 8; ++j) ss = fmaf(s[j], s[j], ss);
  ss = wave_reduce_add(ss);
  float flag = (counts[wave] > 0.0f) ? 1.0f : 0.0f;
  float inv = flag / fmaxf(sqrtf(ss), L2EPS);

  float bc[8], dot = 0.f;
  #pragma unroll
  for (int j = 0; j < 8; ++j) {
    bc[j] = s[j] * inv;                 // zero when flag==0
    dot = fmaf(m[j], bc[j], dot);
  }
  float simi = wave_reduce_add(dot);
  float w = 1.0f - (1.0f - simi) * flag;

  float nm[8], nn = 0.f;
  #pragma unroll
  for (int j = 0; j < 8; ++j) {
    nm[j] = fmaf(w, m[j], (1.0f - w) * bc[j]);
    nn = fmaf(nm[j], nm[j], nn);
  }
  nn = wave_reduce_add(nn);
  float inv2 = 1.0f / fmaxf(sqrtf(nn), L2EPS);

  v8h h;
  #pragma unroll
  for (int j = 0; j < 8; ++j) h[j] = (_Float16)(nm[j] * inv2);
  *(v8h*)out = h;
}

// ---------------------------------------------------------------------------
// Pass C helpers: B-tile register load and WMMA + online-softmax update.
// ---------------------------------------------------------------------------
__device__ __forceinline__ void load_btile(AF (&b)[8], const uint4* __restrict__ Bp) {
  #pragma unroll
  for (int kt = 0; kt < 8; ++kt) {
    b[kt].u[0] = Bp[kt * 4];
    b[kt].u[1] = Bp[kt * 4 + 1];
  }
}

__device__ __forceinline__ void compute_tile(const AF (&a)[8], const AF (&b)[8],
                                             int col, int C, const int (&lab)[8],
                                             float (&s)[8], float (&lv)[8]) {
  v8f acc = {};
  #pragma unroll
  for (int kt = 0; kt < 8; ++kt)
    acc = __builtin_amdgcn_wmma_f32_16x16x32_f16(
        false, a[kt].h, false, b[kt].h, (short)0, acc, false, false);
  float valid = (col < C) ? 1.0f : 0.0f;
  #pragma unroll
  for (int r = 0; r < 8; ++r) {
    float v = acc[r];
    s[r] += valid * __expf(v - 1.0f);
    lv[r] = (col == lab[r]) ? v : lv[r];   // labels < C, pad cols never match
  }
}

// ---------------------------------------------------------------------------
// Pass C: fused GEMM (v_wmma_f32_16x16x32_f16) + softmax-CE reduction.
// One wave per 16-row tile of N. Both operands are unit vectors, so logits
// lie in [-1,1]; fixed-shift exp(v-1) -> logsumexp = 1 + log(sum), no
// running max. B tiles are double-buffered in registers so the 16 b128
// prefetch loads of tile ct+1 overlap the 8 WMMAs + softmax VALU of tile ct.
// ---------------------------------------------------------------------------
__global__ __launch_bounds__(256)
void k_gemm_softmax_loss(const _Float16* __restrict__ featn_h,
                         const _Float16* __restrict__ newmem_h,
                         const int* __restrict__ label,
                         float* __restrict__ out,
                         int N, int C, int Ctiles, float invN) {
  int wave = (int)((blockIdx.x * blockDim.x + threadIdx.x) >> 5);
  int lane = threadIdx.x & 31;
  int m_base = wave * 16;
  if (m_base >= N) return;

  int half = lane >> 4;    // 0: rows 0..7 / K-low ; 1: rows 8..15 / K-high
  int l16  = lane & 15;

  // Preload 8 A fragments (K = 256 = 8 x 32). Contiguous K order (matches B):
  // lane row = l&15; half 0 holds K {k0..k0+15}, half 1 holds K {k0+16..k0+31}.
  const uint4* Arow = (const uint4*)(featn_h + (size_t)(m_base + l16) * DIM);
  AF a[8];
  #pragma unroll
  for (int kt = 0; kt < 8; ++kt) {
    a[kt].u[0] = Arow[kt * 4 + 2 * half];
    a[kt].u[1] = Arow[kt * 4 + 2 * half + 1];
  }

  int lab[8];
  #pragma unroll
  for (int r = 0; r < 8; ++r) lab[r] = label[m_base + half * 8 + r];

  float s[8], lv[8];
  #pragma unroll
  for (int r = 0; r < 8; ++r) { s[r] = 0.0f; lv[r] = 0.0f; }

  // Per-lane B base: column l16, K window starting at 16*half; advance by a
  // constant 16 columns (16*DIM*2 bytes = 512 uint4) per tile.
  const uint4* Bp = (const uint4*)(newmem_h + (size_t)l16 * DIM) + half * 2;
  const int stepB = 16 * DIM / 8;   // uint4 elements per 16-column tile

  AF b0[8], b1[8];
  load_btile(b0, Bp);
  int ct = 0;
  while (true) {
    if (ct + 1 < Ctiles) load_btile(b1, Bp + stepB);
    compute_tile(a, b0, ct * 16 + l16, C, lab, s, lv);
    ++ct; Bp += stepB;
    if (ct >= Ctiles) break;

    if (ct + 1 < Ctiles) load_btile(b0, Bp + stepB);
    compute_tile(a, b1, ct * 16 + l16, C, lab, s, lv);
    ++ct; Bp += stepB;
    if (ct >= Ctiles) break;
  }

  // Butterfly sum over the 16 lanes of each column group.
  #pragma unroll
  for (int r = 0; r < 8; ++r) {
    #pragma unroll
    for (int off = 8; off > 0; off >>= 1) {
      s[r]  += __shfl_xor(s[r], off, 16);
      lv[r] += __shfl_xor(lv[r], off, 16);
    }
  }

  // loss rows: lanes 0..15 hold rows 0..7 totals, lanes 16..31 rows 8..15.
  float part = 0.0f;
  #pragma unroll
  for (int r = 0; r < 8; ++r)
    part += 1.0f + __logf(s[r]) - lv[r];     // logsumexp - x_label
  part += __shfl_xor(part, 16, 32);          // fold the two half-waves
  if (lane == 0) atomicAdd(out, part * invN);
}

// ---------------------------------------------------------------------------
extern "C" void kernel_launch(void* const* d_in, const int* in_sizes, int n_in,
                              void* d_out, int out_size, void* d_ws, size_t ws_size,
                              hipStream_t stream) {
  const float* feat   = (const float*)d_in[0];
  const float* memory = (const float*)d_in[1];
  const int*   label  = (const int*)d_in[2];

  int N = in_sizes[0] / DIM;           // 131072
  int C = in_sizes[1] / DIM;           // 1000
  int Ctiles = (C + 15) / 16;          // 63
  int Cpad = Ctiles * 16;              // 1008
  int Mtiles = (N + 15) / 16;          // 8192

  // Workspace layout (256B aligned)
  char* ws = (char*)d_ws;
  size_t off = 0;
  float* sums = (float*)(ws + off);
  off += (size_t)Cpad * DIM * sizeof(float);
  float* counts = (float*)(ws + off);
  off += ((size_t)Cpad * sizeof(float) + 255) & ~(size_t)255;
  size_t zero_bytes = off;                       // sums + counts region
  _Float16* featn_h = (_Float16*)(ws + off);
  off += (size_t)N * DIM * sizeof(_Float16);
  _Float16* newmem_h = (_Float16*)(ws + off);

  hipMemsetAsync(ws, 0, zero_bytes, stream);
  hipMemsetAsync(d_out, 0, (size_t)out_size * sizeof(float), stream);

  // Pass A: one wave per feat row
  {
    int blocks = (N * 32 + 255) / 256;
    k_normalize_scatter<<<blocks, 256, 0, stream>>>(
        feat, label, featn_h, sums, counts, N);
  }
  // Pass B: one wave per (padded) class row
  {
    int blocks = (Cpad * 32 + 255) / 256;
    k_center_update<<<blocks, 256, 0, stream>>>(
        memory, sums, counts, newmem_h, C, Cpad);
  }
  // Pass C: one wave per 16-row tile
  {
    int blocks = (Mtiles * 32 + 255) / 256;
    k_gemm_softmax_loss<<<blocks, 256, 0, stream>>>(
        featn_h, newmem_h, label, (float*)d_out, N, C, Ctiles,
        1.0f / (float)N);
  }
}
